// ClassFeatureMemoryBank_55800215109814
// MI455X (gfx1250) — matrix operand order repaired and checked by reading, not compiled, not run on hardware
//
#include <hip/hip_runtime.h>
#include <hip/hip_bf16.h>

typedef _Float16 v16h __attribute__((ext_vector_type(16)));
typedef _Float16 v8h  __attribute__((ext_vector_type(8)));
typedef float    v8f  __attribute__((ext_vector_type(8)));
typedef float    v4f  __attribute__((ext_vector_type(4)));

#define N_ROWS   1000000
#define FEAT     128
#define NCLS     150
#define CT       10      // class tiles of 16 (padded to 160)
#define KC       4       // K chunks of 32
#define LDSK     136     // padded K stride in f16 elems (16B-aligned rows, conflict-free)
#define NBLK     1024
#define TEMP     0.15f

__device__ __forceinline__ v16h loadB16(const _Float16* p) {
    v8h x0 = *(const v8h*)p;
    v8h x1 = *(const v8h*)(p + 8);
    return __builtin_shufflevector(x0, x1, 0,1,2,3,4,5,6,7,8,9,10,11,12,13,14,15);
}

__global__ __launch_bounds__(256)
__attribute__((amdgpu_num_vgpr(256)))   // stay in the low VGPR bank: no s_set_vgpr_msb churn
void proto_ce_main(
    const float* __restrict__ feat, const int* __restrict__ labels,
    const float* __restrict__ proto, float* __restrict__ ws)
{
    extern __shared__ char smemRaw[];
    _Float16* sHi = (_Float16*)smemRaw;           // [160][LDSK]
    _Float16* sLo = sHi + 160 * LDSK;             // [160][LDSK]
    float*  slots = (float*)(sLo + 160 * LDSK);   // [8]

    const int tid = threadIdx.x;

    // ---- stage prototypes once per block: fp32 -> f16 hi + f16 residual ----
    for (int i = tid; i < 160 * FEAT; i += 256) {
        int row = i >> 7, col = i & 127;
        float v = (row < NCLS) ? proto[row * FEAT + col] : 0.0f;
        _Float16 h = (_Float16)v;
        sHi[row * LDSK + col] = h;
        sLo[row * LDSK + col] = (_Float16)(v - (float)h);
    }
    __syncthreads();

    const int lane = tid & 31;
    const int wave = tid >> 5;
    const int half = lane >> 4;
    const int l15  = lane & 15;

    float blockLoss = 0.0f;  // per-lane partial (only l15==0 lanes accumulate)
    const v8f vzero = {};

    const int totalGroups = N_ROWS / 16;  // 62500, exact
    for (int g = blockIdx.x * 8 + wave; g < totalGroups; g += gridDim.x * 8) {
        const int rowBase = g * 16;
        const float* rp = feat + (size_t)(rowBase + l15) * FEAT;

        float sumsq = 0.f;
        v8f acc[CT];
        #pragma unroll
        for (int t = 0; t < CT; ++t) acc[t] = vzero;

        // ---- chunk-local A fragments; manual B double-buffer ----
        #pragma unroll
        for (int c = 0; c < KC; ++c) {
            const int kb = c * 32 + half * 16;       // B layout: K = c*32 + half*16 + idx
            const _Float16* colH = sHi + l15 * LDSK + kb;   // + t*16*LDSK per tile
            const _Float16* colL = sLo + l15 * LDSK + kb;

            // issue tile-0 B loads early: LDS latency hidden under loads+conversion below
            v16h bHi = loadB16(colH);
            v16h bLo = loadB16(colL);

            // this chunk's feature strips (A layout: K = c*32 + half*16 + idx)
            const int k0 = c * 32 + half * 16;
            v4f f0 = *(const v4f*)(rp + k0);
            v4f f1 = *(const v4f*)(rp + k0 + 4);
            v4f f2 = *(const v4f*)(rp + k0 + 8);
            v4f f3 = *(const v4f*)(rp + k0 + 12);

            // convert to f16 hi/lo A fragments, accumulate sumsq
            v16h aHi, aLo;
            #pragma unroll
            for (int j = 0; j < 4; ++j) {
                float v0 = f0[j], v1 = f1[j], v2 = f2[j], v3 = f3[j];
                sumsq += v0 * v0 + v1 * v1 + v2 * v2 + v3 * v3;
                _Float16 h;
                h = (_Float16)v0; aHi[j]      = h; aLo[j]      = (_Float16)(v0 - (float)h);
                h = (_Float16)v1; aHi[4 + j]  = h; aLo[4 + j]  = (_Float16)(v1 - (float)h);
                h = (_Float16)v2; aHi[8 + j]  = h; aLo[8 + j]  = (_Float16)(v2 - (float)h);
                h = (_Float16)v3; aHi[12 + j] = h; aLo[12 + j] = (_Float16)(v3 - (float)h);
            }

            // 10 class tiles, B fragments double-buffered one tile ahead
            #pragma unroll
            for (int t = 0; t < CT; ++t) {
                v16h bHiN, bLoN;
                if (t + 1 < CT) {
                    bHiN = loadB16(colH + (t + 1) * 16 * LDSK);
                    bLoN = loadB16(colL + (t + 1) * 16 * LDSK);
                } else {
                    bHiN = bHi; bLoN = bLo;
                }
                acc[t] = __builtin_amdgcn_wmma_f32_16x16x32_f16(false, aHi, false, bHi, (short)0, acc[t], false, false);
                acc[t] = __builtin_amdgcn_wmma_f32_16x16x32_f16(false, aLo, false, bHi, (short)0, acc[t], false, false);
                acc[t] = __builtin_amdgcn_wmma_f32_16x16x32_f16(false, aHi, false, bLo, (short)0, acc[t], false, false);
                bHi = bHiN; bLo = bLoN;
            }

            // keep live ranges chunk-local; stop scheduler from cross-chunk hoisting
            __builtin_amdgcn_sched_barrier(0);
        }

        // lanes L and L^16 cover the same row; combine to full row norm
        sumsq += __shfl_xor(sumsq, 16, 32);
        float scaleAll = 1.0f / (fmaxf(sqrtf(sumsq), 1e-12f) * TEMP);
        int   labAll   = labels[rowBase + l15];

        // ---- per-row scale, logsumexp, CE ----  (C/D layout: M = r + 8*half, N = lane&15)
        #pragma unroll
        for (int r = 0; r < 8; ++r) {
            const int srcLane = r + 8 * half;                 // lane owning this row's norm/label
            float s  = __shfl(scaleAll, srcLane, 32);
            int  lab = __shfl(labAll,  srcLane, 32);
            float m = -__builtin_inff(), pick = 0.f;
            float lg[CT];
            #pragma unroll
            for (int t = 0; t < CT; ++t) {
                float l = acc[t][r] * s;
                if (t == CT - 1 && l15 >= (NCLS - 16 * (CT - 1))) l = -__builtin_inff();  // pad classes
                lg[t] = l;
                m = fmaxf(m, l);
                pick += (t * 16 + l15 == lab) ? l : 0.f;
            }
            m = fmaxf(m, __shfl_xor(m, 1, 32));
            m = fmaxf(m, __shfl_xor(m, 2, 32));
            m = fmaxf(m, __shfl_xor(m, 4, 32));
            m = fmaxf(m, __shfl_xor(m, 8, 32));
            float e = 0.f;
            #pragma unroll
            for (int t = 0; t < CT; ++t) e += __expf(lg[t] - m);   // v_exp_f32; arg in [-inf,0]
            e += __shfl_xor(e, 1, 32);
            e += __shfl_xor(e, 2, 32);
            e += __shfl_xor(e, 4, 32);
            e += __shfl_xor(e, 8, 32);
            pick += __shfl_xor(pick, 1, 32);
            pick += __shfl_xor(pick, 2, 32);
            pick += __shfl_xor(pick, 4, 32);
            pick += __shfl_xor(pick, 8, 32);
            float loss = (m + __logf(e)) - pick;                   // e in [1, 150]
            if (l15 == 0) blockLoss += loss;    // one lane per half-row-set accumulates
        }
    }

    // ---- deterministic block reduction ----
    blockLoss += __shfl_xor(blockLoss, 16, 32);
    if (lane == 0) slots[wave] = blockLoss;
    __syncthreads();
    if (tid == 0) {
        float s = 0.f;
        #pragma unroll
        for (int w = 0; w < 8; ++w) s += slots[w];
        ws[blockIdx.x] = s;
    }
}

__global__ void proto_ce_reduce(const float* __restrict__ ws, float* __restrict__ out)
{
    if (threadIdx.x == 0 && blockIdx.x == 0) {
        double s = 0.0;
        for (int i = 0; i < NBLK; ++i) s += (double)ws[i];
        out[0] = (float)(s / (double)N_ROWS);
    }
}

extern "C" void kernel_launch(void* const* d_in, const int* in_sizes, int n_in,
                              void* d_out, int out_size, void* d_ws, size_t ws_size,
                              hipStream_t stream) {
    const float* feat   = (const float*)d_in[0];
    const int*   labels = (const int*)d_in[1];
    const float* proto  = (const float*)d_in[2];
    float* out = (float*)d_out;
    float* ws  = (float*)d_ws;

    size_t smem = (size_t)(160 * LDSK) * 2 * sizeof(_Float16) + 8 * sizeof(float);
    proto_ce_main<<<NBLK, 256, smem, stream>>>(feat, labels, proto, ws);
    proto_ce_reduce<<<1, 32, 0, stream>>>(ws, out);
}
